// LigerFusedLinearSimPOLoss_54966991454878
// MI455X (gfx1250) — compile-verified, matrix operand order checked by reading, not run on hardware
//
#include <hip/hip_runtime.h>
#include <hip/hip_bf16.h>

// ---------------------------------------------------------------------------
// LigerFusedLinearSimPOLoss for MI455X (gfx1250, wave32, WMMA bf16 + TDM)
//   B=8, T=512, H=2048, V=32000; BETA=0.1, GAMMA=0.5, IGNORE=-100
// Pipeline:
//   1) cvt: fp32 -> bf16 for input (16.8MB) and weight (131MB): whole bf16
//      working set (148MB) fits in the 192MB L2.
//   2) gemm+online-softmax: grid (64 token tiles x 10 vocab segments).
//      The full 64x2048 bf16 A tile (256KB) is loaded ONCE per workgroup into
//      LDS by a single 2D tensor_load_to_lds (TDM), then the whole vocab sweep
//      runs barrier-free: ds_load_b128 A frags + pipelined global B frags +
//      v_wmma_f32_16x16x32_bf16, maintaining per-row running (max, sumexp).
//   3) finalize: merge segment partials -> logZ, target logit via fp32 dot,
//      masked per-sequence average.
//   4) loss: SimPO pairwise softplus on 4 pairs -> scalar.
// ---------------------------------------------------------------------------

typedef __attribute__((ext_vector_type(16))) __bf16        v16bf;
typedef __attribute__((ext_vector_type(8)))  float         v8f;
typedef __attribute__((ext_vector_type(4)))  unsigned int  v4u;
typedef __attribute__((ext_vector_type(8)))  int           v8i;
typedef __attribute__((ext_vector_type(4)))  int           v4i;

union BF16Frag {
    uint4 u4[2];
    v16bf v;
};

#define NB      8
#define NT      512
#define NH      2048
#define NV      32000
#define NROWS   (NB * NT)        // 4096
#define M_TILE  64               // tokens per workgroup (A tile rows)
#define SEG     10               // vocab segments
#define VSEG    (NV / SEG)       // 3200
#define CHUNK   128              // vocab per WG iteration (2 N-waves x 64)
#define A_LDS_BYTES (M_TILE * NH * 2)   // 256 KB dynamic LDS

// ---------------------------------------------------------------- conversion
__global__ void cvt_f32_bf16_kernel(const float* __restrict__ src,
                                    __hip_bfloat16* __restrict__ dst,
                                    size_t n) {
    size_t i = (size_t)blockIdx.x * blockDim.x + threadIdx.x;
    size_t stride = (size_t)gridDim.x * blockDim.x;
    for (; i < n; i += stride) dst[i] = __float2bfloat16(src[i]);
}

// ------------------------------------------------- GEMM + online log-softmax
__global__ __launch_bounds__(256) void simpo_gemm_softmax_kernel(
    const __hip_bfloat16* __restrict__ inB,   // [4096][2048] bf16
    const __hip_bfloat16* __restrict__ wB,    // [32000][2048] bf16
    const float* __restrict__ bias,           // [32000]
    float2* __restrict__ stats)               // [4096][SEG] (max, sumexp)
{
    extern __shared__ __align__(16) __hip_bfloat16 As[];   // [M_TILE][NH], 256KB
    __shared__ float sMax[2][M_TILE];
    __shared__ float sSum[2][M_TILE];

    const int tid    = threadIdx.x;
    const int lane   = tid & 31;
    const int wave   = tid >> 5;          // 0..7
    const int wave_m = wave & 3;          // row group of 16
    const int wave_n = wave >> 2;         // 0..1 (N halves of the 128 chunk)
    const int half   = lane >> 4;         // lane half (K-split layout)
    const int ln     = lane & 15;

    const int row_base = blockIdx.x * M_TILE;
    const int seg      = blockIdx.y;
    const int vstart   = seg * VSEG;

    // ---- one TDM 2D tile load: input[row_base : row_base+64][0:2048] -> LDS
    if (wave == 0) {
        const unsigned long long gaddr =
            (unsigned long long)(uintptr_t)(inB + (size_t)row_base * NH);
        const unsigned int laddr = (unsigned int)(uintptr_t)(&As[0]);

        v4u g0;
        g0[0] = 1u;                                   // count=1, not restore
        g0[1] = laddr;                                // lds_addr (bytes)
        g0[2] = (unsigned int)(gaddr & 0xffffffffu);  // global_addr[31:0]
        g0[3] = (unsigned int)((gaddr >> 32) & 0x01ffffffu) | (2u << 30); // type=2

        v8i g1;
        g1[0] = (int)(1u << 16);                      // data_size=1 (2 bytes)
        g1[1] = (int)((NH & 0xffffu) << 16);          // tensor_dim0[15:0] @63:48
        g1[2] = (int)(((NH >> 16) & 0xffffu) |        // tensor_dim0[31:16]
                      ((M_TILE & 0xffffu) << 16));    // tensor_dim1[15:0]
        g1[3] = (int)(((M_TILE >> 16) & 0xffffu) |    // tensor_dim1[31:16]
                      ((unsigned)NH << 16));          // tile_dim0 = 2048
        g1[4] = (int)(M_TILE & 0xffffu);              // tile_dim1 = 64, tile_dim2=0
        g1[5] = (int)NH;                              // tensor_dim0_stride[31:0]
        g1[6] = 0;                                    // stride hi / dim1_stride lo
        g1[7] = 0;

        v4i gz4;
        gz4[0] = 0; gz4[1] = 0; gz4[2] = 0; gz4[3] = 0;
        v8i gz8;
        gz8[0] = 0; gz8[1] = 0; gz8[2] = 0; gz8[3] = 0;
        gz8[4] = 0; gz8[5] = 0; gz8[6] = 0; gz8[7] = 0;

        __builtin_amdgcn_tensor_load_to_lds(g0, g1, gz4, gz4, gz8, 0);
        __builtin_amdgcn_s_wait_tensorcnt(0);
    }
    __syncthreads();   // A tile resident in LDS for the rest of the kernel

    float run_max[8], run_sum[8];
#pragma unroll
    for (int i = 0; i < 8; ++i) { run_max[i] = -3.0e38f; run_sum[i] = 0.0f; }

    const int abase = (wave_m * 16 + ln) * NH + half * 8;   // A frag LDS base

    for (int c = 0; c < VSEG / CHUNK; ++c) {
        const int n0 = vstart + c * CHUNK + wave_n * 64;    // this wave's N base
        v8f acc[4] = {};

        // per-lane B base pointers (one weight row per lane per tile)
        const __hip_bfloat16* bp0 = wB + (size_t)(n0 +  0 + ln) * NH + half * 8;
        const __hip_bfloat16* bp1 = wB + (size_t)(n0 + 16 + ln) * NH + half * 8;
        const __hip_bfloat16* bp2 = wB + (size_t)(n0 + 32 + ln) * NH + half * 8;
        const __hip_bfloat16* bp3 = wB + (size_t)(n0 + 48 + ln) * NH + half * 8;

        BF16Frag ba[4], bb[4];
        // prologue: fetch k=0 fragments into buffer A
        ba[0].u4[0] = *(const uint4*)(bp0);      ba[0].u4[1] = *(const uint4*)(bp0 + 16);
        ba[1].u4[0] = *(const uint4*)(bp1);      ba[1].u4[1] = *(const uint4*)(bp1 + 16);
        ba[2].u4[0] = *(const uint4*)(bp2);      ba[2].u4[1] = *(const uint4*)(bp2 + 16);
        ba[3].u4[0] = *(const uint4*)(bp3);      ba[3].u4[1] = *(const uint4*)(bp3 + 16);

        for (int kk = 0; kk < NH; kk += 64) {
            // prefetch k = kk+32 into buffer B while computing on buffer A
            const int kn = kk + 32;
            bb[0].u4[0] = *(const uint4*)(bp0 + kn); bb[0].u4[1] = *(const uint4*)(bp0 + kn + 16);
            bb[1].u4[0] = *(const uint4*)(bp1 + kn); bb[1].u4[1] = *(const uint4*)(bp1 + kn + 16);
            bb[2].u4[0] = *(const uint4*)(bp2 + kn); bb[2].u4[1] = *(const uint4*)(bp2 + kn + 16);
            bb[3].u4[0] = *(const uint4*)(bp3 + kn); bb[3].u4[1] = *(const uint4*)(bp3 + kn + 16);

            BF16Frag af;
            af.u4[0] = *(const uint4*)(&As[abase + kk]);
            af.u4[1] = *(const uint4*)(&As[abase + kk + 16]);
            acc[0] = __builtin_amdgcn_wmma_f32_16x16x32_bf16(false, af.v, false, ba[0].v, (short)0, acc[0], false, false);
            acc[1] = __builtin_amdgcn_wmma_f32_16x16x32_bf16(false, af.v, false, ba[1].v, (short)0, acc[1], false, false);
            acc[2] = __builtin_amdgcn_wmma_f32_16x16x32_bf16(false, af.v, false, ba[2].v, (short)0, acc[2], false, false);
            acc[3] = __builtin_amdgcn_wmma_f32_16x16x32_bf16(false, af.v, false, ba[3].v, (short)0, acc[3], false, false);

            // prefetch k = kk+64 into buffer A while computing on buffer B
            if (kk + 64 < NH) {
                const int km = kk + 64;
                ba[0].u4[0] = *(const uint4*)(bp0 + km); ba[0].u4[1] = *(const uint4*)(bp0 + km + 16);
                ba[1].u4[0] = *(const uint4*)(bp1 + km); ba[1].u4[1] = *(const uint4*)(bp1 + km + 16);
                ba[2].u4[0] = *(const uint4*)(bp2 + km); ba[2].u4[1] = *(const uint4*)(bp2 + km + 16);
                ba[3].u4[0] = *(const uint4*)(bp3 + km); ba[3].u4[1] = *(const uint4*)(bp3 + km + 16);
            }

            BF16Frag ag;
            ag.u4[0] = *(const uint4*)(&As[abase + kn]);
            ag.u4[1] = *(const uint4*)(&As[abase + kn + 16]);
            acc[0] = __builtin_amdgcn_wmma_f32_16x16x32_bf16(false, ag.v, false, bb[0].v, (short)0, acc[0], false, false);
            acc[1] = __builtin_amdgcn_wmma_f32_16x16x32_bf16(false, ag.v, false, bb[1].v, (short)0, acc[1], false, false);
            acc[2] = __builtin_amdgcn_wmma_f32_16x16x32_bf16(false, ag.v, false, bb[2].v, (short)0, acc[2], false, false);
            acc[3] = __builtin_amdgcn_wmma_f32_16x16x32_bf16(false, ag.v, false, bb[3].v, (short)0, acc[3], false, false);
        }

        // add bias (per N column = per lane per tile)
#pragma unroll
        for (int t = 0; t < 4; ++t) {
            const float bv = bias[n0 + t * 16 + ln];
#pragma unroll
            for (int i = 0; i < 8; ++i) acc[t][i] += bv;
        }

        // online softmax update. C layout: reg i, lane -> row i + 8*half,
        // col = lane&15; reduce over 4 tiles + 16 lanes of each half.
#pragma unroll
        for (int i = 0; i < 8; ++i) {
            float cmax = fmaxf(fmaxf(acc[0][i], acc[1][i]),
                               fmaxf(acc[2][i], acc[3][i]));
            cmax = fmaxf(cmax, __shfl_xor(cmax, 1));
            cmax = fmaxf(cmax, __shfl_xor(cmax, 2));
            cmax = fmaxf(cmax, __shfl_xor(cmax, 4));
            cmax = fmaxf(cmax, __shfl_xor(cmax, 8));
            const float nmax = fmaxf(run_max[i], cmax);
            float csum = __expf(acc[0][i] - nmax) + __expf(acc[1][i] - nmax) +
                         __expf(acc[2][i] - nmax) + __expf(acc[3][i] - nmax);
            csum += __shfl_xor(csum, 1);
            csum += __shfl_xor(csum, 2);
            csum += __shfl_xor(csum, 4);
            csum += __shfl_xor(csum, 8);
            run_sum[i] = run_sum[i] * __expf(run_max[i] - nmax) + csum;
            run_max[i] = nmax;
        }
    }

    // merge the two N-waves per row, write per-segment partials
    if (ln == 0) {
#pragma unroll
        for (int i = 0; i < 8; ++i) {
            const int r = wave_m * 16 + half * 8 + i;
            sMax[wave_n][r] = run_max[i];
            sSum[wave_n][r] = run_sum[i];
        }
    }
    __syncthreads();
    if (tid < M_TILE) {
        const float m0 = sMax[0][tid], m1 = sMax[1][tid];
        const float M  = fmaxf(m0, m1);
        const float S  = sSum[0][tid] * __expf(m0 - M) +
                         sSum[1][tid] * __expf(m1 - M);
        stats[(size_t)(row_base + tid) * SEG + seg] = make_float2(M, S);
    }
}

// ------------------------------------------------------------------ finalize
__global__ __launch_bounds__(256) void simpo_finalize_kernel(
    const float* __restrict__ inp,      // [4096][2048] fp32
    const float* __restrict__ w,        // [32000][2048] fp32
    const int* __restrict__ target,     // [4096]
    const float* __restrict__ bias,     // [32000]
    const float2* __restrict__ stats,   // [4096][SEG]
    float* __restrict__ avg_out)        // [8]
{
    const int b   = blockIdx.x;
    const int tid = threadIdx.x;
    float sum = 0.0f, cnt = 0.0f;

    for (int t = tid; t < NT; t += 256) {
        const int row = b * NT + t;
        float M = -3.0e38f;
#pragma unroll
        for (int s = 0; s < SEG; ++s)
            M = fmaxf(M, stats[(size_t)row * SEG + s].x);
        float Z = 0.0f;
#pragma unroll
        for (int s = 0; s < SEG; ++s) {
            const float2 p = stats[(size_t)row * SEG + s];
            Z += p.y * __expf(p.x - M);
        }
        const float logZ = M + __logf(Z);

        const int tg = target[row];
        if (tg != -100) {
            const float4* xr = (const float4*)(inp + (size_t)row * NH);
            const float4* wr = (const float4*)(w + (size_t)tg * NH);
            float dot = 0.0f;
            for (int h = 0; h < NH / 4; ++h) {
                const float4 a = xr[h], c = wr[h];
                dot += a.x * c.x + a.y * c.y + a.z * c.z + a.w * c.w;
            }
            sum += (dot + bias[tg]) - logZ;
            cnt += 1.0f;
        }
    }

    __shared__ float rsum[256];
    __shared__ float rcnt[256];
    rsum[tid] = sum; rcnt[tid] = cnt;
    __syncthreads();
    for (int off = 128; off > 0; off >>= 1) {
        if (tid < off) { rsum[tid] += rsum[tid + off]; rcnt[tid] += rcnt[tid + off]; }
        __syncthreads();
    }
    if (tid == 0) avg_out[b] = rsum[0] / fmaxf(rcnt[0], 1.0f);
}

// --------------------------------------------------------------------- loss
__global__ void simpo_loss_kernel(const float* __restrict__ avg,
                                  float* __restrict__ out) {
    if (threadIdx.x == 0 && blockIdx.x == 0) {
        float loss = 0.0f;
        for (int p = 0; p < NB / 2; ++p) {
            const float d = 0.1f * (avg[p] - avg[p + NB / 2]) - 0.5f;
            // -log_sigmoid(d) = softplus(-d), numerically stable
            loss += (d > 0.0f) ? log1pf(__expf(-d)) : (-d + log1pf(__expf(d)));
        }
        out[0] = loss / (float)(NB / 2);
    }
}

// ------------------------------------------------------------------- launch
extern "C" void kernel_launch(void* const* d_in, const int* in_sizes, int n_in,
                              void* d_out, int out_size, void* d_ws, size_t ws_size,
                              hipStream_t stream) {
    const float* w    = (const float*)d_in[0];   // (V, H)
    const float* inp  = (const float*)d_in[1];   // (B, T, H)
    const int*   tgt  = (const int*)d_in[2];     // (B, T)
    const float* bias = (const float*)d_in[3];   // (V,)

    char* ws = (char*)d_ws;
    const size_t szInB  = (size_t)NROWS * NH * sizeof(__hip_bfloat16);   // 16.8 MB
    const size_t szWB   = (size_t)NV * NH * sizeof(__hip_bfloat16);      // 131 MB
    const size_t szStat = (size_t)NROWS * SEG * sizeof(float2);          // 328 KB
    __hip_bfloat16* inB = (__hip_bfloat16*)ws;
    __hip_bfloat16* wB  = (__hip_bfloat16*)(ws + szInB);
    float2* stats       = (float2*)(ws + szInB + szWB);
    float*  avg         = (float*)(ws + szInB + szWB + szStat);

    cvt_f32_bf16_kernel<<<4096, 256, 0, stream>>>(inp, inB, (size_t)NROWS * NH);
    cvt_f32_bf16_kernel<<<8192, 256, 0, stream>>>(w, wB, (size_t)NV * NH);

    dim3 grid(NROWS / M_TILE, SEG);   // 64 x 10
    simpo_gemm_softmax_kernel<<<grid, 256, A_LDS_BYTES, stream>>>(inB, wB, bias, stats);

    simpo_finalize_kernel<<<NB, 256, 0, stream>>>(inp, w, tgt, bias, stats, avg);
    simpo_loss_kernel<<<1, 32, 0, stream>>>(avg, (float*)d_out);
}
